// norm_attention_893353198104
// MI455X (gfx1250) — compile-verified, matrix operand order checked by reading, not run on hardware
//
#include <hip/hip_runtime.h>

typedef __bf16 v16bf __attribute__((ext_vector_type(16)));
typedef float  v8f   __attribute__((ext_vector_type(8)));

#define DIMC 64
#define GC 8
#define CMID 32
#define NO2 392          // 49 taps * 8 groups
#define NO2P 400         // padded to 25 N-tiles of 16
#define H 128
#define W 128
#define HW (H*W)
#define WT_STRIDE 130    // padded pixel stride (odd dword stride -> no LDS bank conflicts)

__device__ __forceinline__ float bf2f(unsigned short s) {
  unsigned u = ((unsigned)s) << 16;
  return __builtin_bit_cast(float, u);
}
// K-pair base for 16-bit A/B fragment VGPR v, lane-half hi (16x16x32 bf16 layout)
__device__ __forceinline__ int klocal(int v, int hi) {
  return ((v < 4) ? (2 * v) : (8 + 2 * v)) + 8 * hi;
}

__global__ __launch_bounds__(256) void involution_fused(
    const float* __restrict__ x, const float* __restrict__ y,
    const float* __restrict__ w1, const float* __restrict__ b1,
    const float* __restrict__ w2, const float* __restrict__ b2,
    float* __restrict__ out)
{
  __shared__ unsigned       w1p[CMID * DIMC / 2];       //  4 KB  bf16 W1 [n][k] pair-packed
  __shared__ unsigned       w2p[NO2P * CMID / 2];       // 25 KB  bf16 W2 [o2][k] pair-packed
  __shared__ float          mids[8 * 16 * CMID];        // 16 KB  per-wave mid scratch
  __shared__ unsigned short wts[NO2P * WT_STRIDE];      //104 KB  bf16 weight [o2][pixel]

  const int tid = threadIdx.x;
  const int bi  = blockIdx.x;
  const int b   = bi >> 7;      // image
  const int r   = bi & 127;     // row

  // ---- preload W1 / W2 as bf16 via hardware f32->bf16 convert (coalesced) ----
  __bf16* w1s = (__bf16*)w1p;
  for (int i = tid; i < CMID * DIMC; i += 256) w1s[i] = (__bf16)w1[i];
  __bf16* w2s = (__bf16*)w2p;
  for (int i = tid; i < NO2P * CMID; i += 256) {
    float v = (i < NO2 * CMID) ? w2[i] : 0.0f;
    w2s[i] = (__bf16)v;
  }
  __syncthreads();

  const int lane = tid & 31;
  const int wv   = tid >> 5;        // wave id: pixels wv*16 .. wv*16+15
  const int m16  = lane & 15;
  const int hi   = lane >> 4;
  const int pxb  = wv * 16;

  const float* xb = x + (size_t)b * DIMC * HW + (size_t)r * W + pxb + m16;

  // ---- GEMM1: mid(16x32) = X(16x64) * W1^T, bf16 WMMA, f32 acc ----
  v8f acc0 = {}; v8f acc1 = {};
  #pragma unroll
  for (int ks = 0; ks < 2; ++ks) {
    v16bf afrag;
    #pragma unroll
    for (int v = 0; v < 8; ++v) {
      int k = ks * 32 + klocal(v, hi);
      float f0 = xb[(size_t)k * HW];          // coalesced: lanes 0-15 adjacent pixels
      float f1 = xb[(size_t)(k + 1) * HW];
      afrag[2 * v]     = (__bf16)f0;          // v_cvt pk bf16 path
      afrag[2 * v + 1] = (__bf16)f1;
    }
    #pragma unroll
    for (int nt = 0; nt < 2; ++nt) {
      int n = nt * 16 + m16;
      v16bf bfrag;
      const __bf16* wp = (const __bf16*)w1p;
      #pragma unroll
      for (int v = 0; v < 8; ++v) {
        int k = ks * 32 + klocal(v, hi);
        // pair (k, k+1) contiguous -> compiler merges to b32/b64 LDS reads
        bfrag[2 * v]     = wp[n * DIMC + k];
        bfrag[2 * v + 1] = wp[n * DIMC + k + 1];
      }
      if (nt == 0)
        acc0 = __builtin_amdgcn_wmma_f32_16x16x32_bf16(false, afrag, false, bfrag,
                                                       (short)0, acc0, false, false);
      else
        acc1 = __builtin_amdgcn_wmma_f32_16x16x32_bf16(false, afrag, false, bfrag,
                                                       (short)0, acc1, false, false);
    }
  }
  // bias + stash mid (C/D layout -> LDS, same wave reads back in A layout)
  {
    float bias0 = b1[m16];
    float bias1 = b1[16 + m16];
    float* mr = mids + wv * 512;
    #pragma unroll
    for (int j = 0; j < 8; ++j) {
      int m = j + 8 * hi;
      mr[m * CMID + m16]      = acc0[j] + bias0;
      mr[m * CMID + 16 + m16] = acc1[j] + bias1;
    }
  }
  asm volatile("s_wait_dscnt 0" ::: "memory");   // same-wave DS in-order; be explicit

  v16bf a2;
  {
    const float* mr = mids + wv * 512 + m16 * CMID;
    #pragma unroll
    for (int v = 0; v < 8; ++v) {
      int k = klocal(v, hi);
      a2[2 * v]     = (__bf16)mr[k];
      a2[2 * v + 1] = (__bf16)mr[k + 1];
    }
  }

  // ---- GEMM2: weight(16x400) = mid(16x32) * W2^T, 25 WMMAs ----
  __bf16* wtsb = (__bf16*)wts;
  #pragma unroll 1
  for (int nt = 0; nt < 25; ++nt) {
    int o2 = nt * 16 + m16;
    v16bf bfrag;
    const __bf16* wp = (const __bf16*)w2p;
    #pragma unroll
    for (int v = 0; v < 8; ++v) {
      int k = klocal(v, hi);
      bfrag[2 * v]     = wp[o2 * CMID + k];
      bfrag[2 * v + 1] = wp[o2 * CMID + k + 1];
    }
    v8f z = {};
    v8f d = __builtin_amdgcn_wmma_f32_16x16x32_bf16(false, a2, false, bfrag,
                                                    (short)0, z, false, false);
    float bias = (o2 < NO2) ? b2[o2] : 0.0f;
    #pragma unroll
    for (int j = 0; j < 8; ++j) {
      int p = pxb + j + 8 * hi;
      wtsb[o2 * WT_STRIDE + p] = (__bf16)(d[j] + bias);
    }
  }
  __syncthreads();

  // ---- Phase B: out[p,c] = sum_t wt[g(c)*49+t] * y[p+off(t), c] ----
  const int p  = tid & 127;                 // lane-contiguous pixel -> coalesced y
  const int cb = (tid >> 7) << 5;           // channel base 0 or 32
  const float* yb = y + (size_t)b * DIMC * HW;
  #pragma unroll 1
  for (int gi = 0; gi < 4; ++gi) {
    int g = (cb >> 3) + gi;
    float wreg[49];                         // hoist weights, reused by 8 channels
    #pragma unroll
    for (int t = 0; t < 49; ++t)
      wreg[t] = bf2f(wts[(g * 49 + t) * WT_STRIDE + p]);
    #pragma unroll 1
    for (int cc = 0; cc < GC; ++cc) {
      int c = g * GC + cc;
      const float* yc = yb + (size_t)c * HW;
      float acc = 0.0f;
      #pragma unroll
      for (int dy = -3; dy <= 3; ++dy) {
        int yy = r + dy;
        bool rowok = ((unsigned)yy) < (unsigned)H;
        const float* yr = yc + yy * W;
        #pragma unroll
        for (int dx = -3; dx <= 3; ++dx) {
          int xx = p + dx;
          float yv = (rowok && ((unsigned)xx) < (unsigned)W) ? yr[xx] : 0.0f;
          acc = fmaf(wreg[(dy + 3) * 7 + (dx + 3)], yv, acc);
        }
      }
      out[((size_t)(b * DIMC + c) * H + r) * W + p] = acc;
    }
  }
}

extern "C" void kernel_launch(void* const* d_in, const int* in_sizes, int n_in,
                              void* d_out, int out_size, void* d_ws, size_t ws_size,
                              hipStream_t stream) {
  const float* x  = (const float*)d_in[0];
  const float* y  = (const float*)d_in[1];
  const float* w1 = (const float*)d_in[2];
  const float* b1 = (const float*)d_in[3];
  const float* w2 = (const float*)d_in[4];
  const float* b2 = (const float*)d_in[5];
  float* out = (float*)d_out;
  (void)in_sizes; (void)n_in; (void)out_size; (void)d_ws; (void)ws_size;
  involution_fused<<<dim3(4 * 128), dim3(256), 0, stream>>>(x, y, w1, b1, w2, b2, out);
}
